// TabR_32409823216327
// MI455X (gfx1250) — compile-verified, matrix operand order checked by reading, not run on hardware
//
#include <hip/hip_runtime.h>
#include <hip/hip_bf16.h>

typedef __bf16 bf16;
typedef __attribute__((ext_vector_type(16))) __bf16 v16bf;
typedef __attribute__((ext_vector_type(8)))  float  v8f;

#define NCAND 50000
#define BQ    512
#define CTX   96
#define DM    512
#define DB    1024
#define DIN   820
#define DINP  832

#define LDK 48   // padded LDS row stride (elems): 96B, 16B-aligned, bank-spread

#if __has_builtin(__builtin_amdgcn_global_load_async_to_lds_b128)
#define HAVE_ASYNC_LDS 1
typedef int int4x __attribute__((vector_size(16)));
typedef __attribute__((address_space(1))) int4x* as1_int4p;
typedef __attribute__((address_space(3))) int4x* as3_int4p;
#else
#define HAVE_ASYNC_LDS 0
#endif

#if __has_builtin(__builtin_amdgcn_s_wait_asynccnt)
#define WAIT_ASYNC(n) __builtin_amdgcn_s_wait_asynccnt(n)
#else
#define WAIT_ASYNC(n) asm volatile("s_wait_asynccnt %0" :: "i"(n) : "memory")
#endif

// ---------------------------------------------------------------------------
// bf16 WMMA GEMM:  C[M,N] = act(A[M,K] @ B^T + bias) (+res)
// A row-major bf16 (lda=K). B stored (N,K) row-major, ldb = K-stride.
// Block tile 128x128, K-step 32, 256 threads = 8 waves, wave tile 32x64.
// Double-buffered LDS; staging via GLOBAL_LOAD_ASYNC_TO_LDS (ASYNCcnt
// pipelined) when available. Out-of-range rows/cols are clamped at load
// (their products land only in discarded output rows/cols).
// ---------------------------------------------------------------------------
__global__ __launch_bounds__(256)
void gemm_bf16_kernel(const bf16* __restrict__ A, const bf16* __restrict__ B,
                      const float* __restrict__ bias, const float* __restrict__ res,
                      float* __restrict__ Cf, bf16* __restrict__ Cb,
                      int M, int N, int K, int ldb, int relu)
{
    __shared__ bf16 lA[2][128][LDK];
    __shared__ bf16 lB[2][128][LDK];

    const int tid  = threadIdx.x;
    const int lane = tid & 31;
    const int wave = tid >> 5;
    const int wrow = wave & 3;   // M sub-tile *32
    const int wcol = wave >> 2;  // N sub-tile *64
    const int bm = blockIdx.y * 128;
    const int bn = blockIdx.x * 128;

    const int lm = lane & 15;
    const int hi = lane >> 4;
    const int kb = hi * 8;

    v8f acc[2][4] = {};

    // staging: 128 rows x 32 elems = 512 16B-chunks per matrix; 2 chunks/thread
    const int crow = tid >> 2;         // 0..63 (+64 for second chunk)
    const int cks  = (tid & 3) * 8;    // 0,8,16,24

    auto stage = [&](int buf, int k0) {
#pragma unroll
        for (int l = 0; l < 2; ++l) {
            const int row = l * 64 + crow;
            const int gm = min(bm + row, M - 1);
            const bf16* gp = A + (size_t)gm * K + k0 + cks;
#if HAVE_ASYNC_LDS
            __builtin_amdgcn_global_load_async_to_lds_b128(
                (as1_int4p)gp, (as3_int4p)&lA[buf][row][cks], 0, 0);
#else
            *(uint4*)&lA[buf][row][cks] = *(const uint4*)gp;
#endif
        }
#pragma unroll
        for (int l = 0; l < 2; ++l) {
            const int row = l * 64 + crow;
            const int gn = min(bn + row, N - 1);
            const bf16* gp = B + (size_t)gn * ldb + k0 + cks;
#if HAVE_ASYNC_LDS
            __builtin_amdgcn_global_load_async_to_lds_b128(
                (as1_int4p)gp, (as3_int4p)&lB[buf][row][cks], 0, 0);
#else
            *(uint4*)&lB[buf][row][cks] = *(const uint4*)gp;
#endif
        }
    };

    int cur = 0;
    stage(0, 0);
    for (int k0 = 0; k0 < K; k0 += 32) {
        if (k0 + 32 < K) {
            stage(cur ^ 1, k0 + 32);   // prefetch next tile into other buffer
            WAIT_ASYNC(4);             // oldest 4 (current tile) have landed
        } else {
            WAIT_ASYNC(0);
        }
        __syncthreads();

        union Frag { v16bf v; uint4 q[2]; } fa0, fa1, fb;
        const int ar = wrow * 32 + lm;
        fa0.q[0] = *(const uint4*)&lA[cur][ar][kb];
        fa0.q[1] = *(const uint4*)&lA[cur][ar][kb + 16];
        fa1.q[0] = *(const uint4*)&lA[cur][ar + 16][kb];
        fa1.q[1] = *(const uint4*)&lA[cur][ar + 16][kb + 16];
#pragma unroll
        for (int j = 0; j < 4; ++j) {
            const int bc = wcol * 64 + j * 16 + lm;
            fb.q[0] = *(const uint4*)&lB[cur][bc][kb];
            fb.q[1] = *(const uint4*)&lB[cur][bc][kb + 16];
            acc[0][j] = __builtin_amdgcn_wmma_f32_16x16x32_bf16(
                false, fa0.v, false, fb.v, (short)0, acc[0][j], false, false);
            acc[1][j] = __builtin_amdgcn_wmma_f32_16x16x32_bf16(
                false, fa1.v, false, fb.v, (short)0, acc[1][j], false, false);
        }
        __syncthreads();   // all waves done reading buf before it is re-staged
        cur ^= 1;
    }

    // ---- epilogue: bias / residual / relu, f32 and/or bf16 stores ----
#pragma unroll
    for (int j = 0; j < 4; ++j) {
        const int col = bn + wcol * 64 + j * 16 + lm;
        if (col < N) {
            const float bv = bias ? bias[col] : 0.0f;
#pragma unroll
            for (int ai = 0; ai < 2; ++ai) {
#pragma unroll
                for (int i = 0; i < 8; ++i) {
                    const int row = bm + wrow * 32 + ai * 16 + hi * 8 + i;
                    if (row < M) {
                        const size_t o = (size_t)row * N + col;
                        float r = acc[ai][j][i] + bv;
                        if (res)  r += res[o];
                        if (relu) r = r > 0.0f ? r : 0.0f;
                        if (Cf) Cf[o] = r;
                        if (Cb) Cb[o] = (bf16)r;
                    }
                }
            }
        }
    }
}

// ---------------------------------------------------------------------------
// Weight convert+transpose: src (K,N) f32 row-major -> dst (N,Kpad) bf16,
// zero-padding k in [K,Kpad).
// ---------------------------------------------------------------------------
__global__ void cvt_wT_kernel(const float* __restrict__ src, bf16* __restrict__ dst,
                              int K, int N, int Kpad)
{
    int i = blockIdx.x * 256 + threadIdx.x;
    if (i >= N * Kpad) return;
    int n = i / Kpad, k = i - n * Kpad;
    dst[i] = (k < K) ? (bf16)src[(size_t)k * N + n] : (bf16)0.0f;
}

// ---------------------------------------------------------------------------
// PLR feature encoder: row -> bf16 feature vector of width DINP (=832).
// ---------------------------------------------------------------------------
__global__ __launch_bounds__(256)
void plr_kernel(const float* __restrict__ xnum, const float* __restrict__ xcat,
                const float* __restrict__ freq, const float* __restrict__ plw,
                const float* __restrict__ plb, bf16* __restrict__ feat, int rows)
{
    int r = blockIdx.x;
    if (r >= rows) return;
    __shared__ float xr[50];
    if (threadIdx.x < 50) xr[threadIdx.x] = xnum[(size_t)r * 50 + threadIdx.x];
    __syncthreads();
    for (int o = threadIdx.x; o < DINP; o += 256) {
        float v;
        if (o < 800) {
            int f = o >> 4, j = o & 15;
            float s = plb[j];
            float xv = xr[f];
#pragma unroll
            for (int q = 0; q < 16; ++q) {
                float p = 6.28318530718f * freq[f * 16 + q] * xv;
                s += __cosf(p) * plw[q * 16 + j] + __sinf(p) * plw[(16 + q) * 16 + j];
            }
            v = s > 0.0f ? s : 0.0f;
        } else if (o < DIN) {
            v = xcat[(size_t)r * 20 + (o - 800)];
        } else {
            v = 0.0f;
        }
        feat[(size_t)r * DINP + o] = (bf16)v;
    }
}

// ---------------------------------------------------------------------------
// Row LayerNorm (512 cols) -> bf16
// ---------------------------------------------------------------------------
__global__ __launch_bounds__(256)
void ln_kernel(const float* __restrict__ X, const float* __restrict__ g,
               const float* __restrict__ b, bf16* __restrict__ out)
{
    int r = blockIdx.x, tid = threadIdx.x;
    const float* x = X + (size_t)r * DM;
    __shared__ float red[256];
    float v0 = x[tid], v1 = x[tid + 256];
    red[tid] = v0 + v1; __syncthreads();
    for (int o = 128; o > 0; o >>= 1) { if (tid < o) red[tid] += red[tid + o]; __syncthreads(); }
    float mean = red[0] * (1.0f / DM); __syncthreads();
    float d0 = v0 - mean, d1 = v1 - mean;
    red[tid] = d0 * d0 + d1 * d1; __syncthreads();
    for (int o = 128; o > 0; o >>= 1) { if (tid < o) red[tid] += red[tid + o]; __syncthreads(); }
    float rs = rsqrtf(red[0] * (1.0f / DM) + 1e-5f);
    out[(size_t)r * DM + tid]       = (bf16)(d0 * rs * g[tid]       + b[tid]);
    out[(size_t)r * DM + tid + 256] = (bf16)(d1 * rs * g[tid + 256] + b[tid + 256]);
}

// ---------------------------------------------------------------------------
// Row squared norms (512 cols)
// ---------------------------------------------------------------------------
__global__ __launch_bounds__(256)
void rownorm_kernel(const float* __restrict__ X, float* __restrict__ nrm)
{
    int r = blockIdx.x, tid = threadIdx.x;
    const float* x = X + (size_t)r * DM;
    __shared__ float red[256];
    float v0 = x[tid], v1 = x[tid + 256];
    red[tid] = v0 * v0 + v1 * v1; __syncthreads();
    for (int o = 128; o > 0; o >>= 1) { if (tid < o) red[tid] += red[tid + o]; __syncthreads(); }
    if (tid == 0) nrm[r] = red[0];
}

// ---------------------------------------------------------------------------
// Top-96 by L2 distance: d2[j] = cnorm[j] - 2*dot[j] (+const). Destroys dot.
// ---------------------------------------------------------------------------
__global__ __launch_bounds__(256)
void topk_kernel(float* __restrict__ dot, const float* __restrict__ cnorm,
                 int* __restrict__ ctx, int ncand)
{
    int b = blockIdx.x, tid = threadIdx.x;
    float* drow = dot + (size_t)b * ncand;
    __shared__ float bv[256];
    __shared__ int   bi[256];
    for (int c = 0; c < CTX; ++c) {
        float best = 3.4e38f; int besti = 0;
        for (int j = tid; j < ncand; j += 256) {
            float d = cnorm[j] - 2.0f * drow[j];
            if (d < best) { best = d; besti = j; }
        }
        bv[tid] = best; bi[tid] = besti; __syncthreads();
        for (int o = 128; o > 0; o >>= 1) {
            if (tid < o && bv[tid + o] < bv[tid]) { bv[tid] = bv[tid + o]; bi[tid] = bi[tid + o]; }
            __syncthreads();
        }
        if (tid == 0) { ctx[b * CTX + c] = bi[0]; drow[bi[0]] = -1.0e30f; }
        __syncthreads();
    }
}

// ---------------------------------------------------------------------------
// Gather: diff(bf16), values_init = y*lw+lb (f32), sim = -|diff|^2
// ---------------------------------------------------------------------------
__global__ __launch_bounds__(256)
void gather_kernel(const float* __restrict__ qk, const float* __restrict__ ck,
                   const int* __restrict__ ctx, const float* __restrict__ cy,
                   const float* __restrict__ lw, const float* __restrict__ lb,
                   bf16* __restrict__ diff, float* __restrict__ values,
                   float* __restrict__ sim)
{
    int blk = blockIdx.x;               // b*CTX + c
    int b = blk / CTX;
    int idx = ctx[blk];
    int tid = threadIdx.x;
    float y = cy[idx];
    __shared__ float red[256];
    float acc = 0.0f;
    for (int d = tid; d < DM; d += 256) {
        float df = qk[(size_t)b * DM + d] - ck[(size_t)idx * DM + d];
        diff[(size_t)blk * DM + d]   = (bf16)df;
        values[(size_t)blk * DM + d] = y * lw[d] + lb[d];
        acc += df * df;
    }
    red[tid] = acc; __syncthreads();
    for (int o = 128; o > 0; o >>= 1) { if (tid < o) red[tid] += red[tid + o]; __syncthreads(); }
    if (tid == 0) sim[blk] = -red[0];
}

// ---------------------------------------------------------------------------
// Softmax over CTX + weighted value aggregation into qx (residual add)
// ---------------------------------------------------------------------------
__global__ __launch_bounds__(256)
void mix_kernel(float* __restrict__ qx, const float* __restrict__ sim,
                const float* __restrict__ values)
{
    int b = blockIdx.x, tid = threadIdx.x;
    __shared__ float sp[CTX];
    if (tid < CTX) sp[tid] = sim[b * CTX + tid];
    __syncthreads();
    float mx = -3.4e38f;
    for (int c = 0; c < CTX; ++c) mx = fmaxf(mx, sp[c]);
    float sum = 0.0f;
    for (int c = 0; c < CTX; ++c) sum += __expf(sp[c] - mx);
    float inv = 1.0f / sum;
    __syncthreads();
    if (tid < CTX) sp[tid] = __expf(sp[tid] - mx) * inv;   // probs
    __syncthreads();
    for (int d = tid; d < DM; d += 256) {
        float a = qx[(size_t)b * DM + d];
        for (int c = 0; c < CTX; ++c)
            a += sp[c] * values[((size_t)b * CTX + c) * DM + d];
        qx[(size_t)b * DM + d] = a;
    }
}

// ---------------------------------------------------------------------------
// Head: out = relu(LN(x)) @ head_w + head_bias
// ---------------------------------------------------------------------------
__global__ __launch_bounds__(256)
void head_kernel(const float* __restrict__ X, const float* __restrict__ g,
                 const float* __restrict__ bb, const float* __restrict__ hw,
                 const float* __restrict__ hb, float* __restrict__ out)
{
    int b = blockIdx.x, tid = threadIdx.x;
    const float* x = X + (size_t)b * DM;
    __shared__ float red[256];
    float v0 = x[tid], v1 = x[tid + 256];
    red[tid] = v0 + v1; __syncthreads();
    for (int o = 128; o > 0; o >>= 1) { if (tid < o) red[tid] += red[tid + o]; __syncthreads(); }
    float mean = red[0] * (1.0f / DM); __syncthreads();
    float d0 = v0 - mean, d1 = v1 - mean;
    red[tid] = d0 * d0 + d1 * d1; __syncthreads();
    for (int o = 128; o > 0; o >>= 1) { if (tid < o) red[tid] += red[tid + o]; __syncthreads(); }
    float rs = rsqrtf(red[0] * (1.0f / DM) + 1e-5f); __syncthreads();
    float h0 = fmaxf(d0 * rs * g[tid]       + bb[tid],       0.0f);
    float h1 = fmaxf(d1 * rs * g[tid + 256] + bb[tid + 256], 0.0f);
    red[tid] = h0 * hw[tid] + h1 * hw[tid + 256]; __syncthreads();
    for (int o = 128; o > 0; o >>= 1) { if (tid < o) red[tid] += red[tid + o]; __syncthreads(); }
    if (tid == 0) out[b] = red[0] + hb[0];
}

// ---------------------------------------------------------------------------
extern "C" void kernel_launch(void* const* d_in, const int* in_sizes, int n_in,
                              void* d_out, int out_size, void* d_ws, size_t ws_size,
                              hipStream_t stream)
{
    const float* x_num   = (const float*)d_in[0];
    const float* x_cat   = (const float*)d_in[1];
    const float* c_num   = (const float*)d_in[2];
    const float* c_cat   = (const float*)d_in[3];
    const float* c_y     = (const float*)d_in[4];
    /* d_in[5] = context_size (96, hardcoded) */
    const float* freq    = (const float*)d_in[6];
    const float* plr_w   = (const float*)d_in[7];
    const float* plr_b   = (const float*)d_in[8];
    const float* lin_w   = (const float*)d_in[9];
    const float* lin_b   = (const float*)d_in[10];
    const float* enc_w1  = (const float*)d_in[11];
    const float* enc_b1  = (const float*)d_in[12];
    const float* enc_w2  = (const float*)d_in[13];
    const float* enc_b2  = (const float*)d_in[14];
    const float* mix_g   = (const float*)d_in[15];
    const float* mix_b   = (const float*)d_in[16];
    const float* K_w     = (const float*)d_in[17];
    const float* K_b     = (const float*)d_in[18];
    const float* label_w = (const float*)d_in[19];
    const float* label_b = (const float*)d_in[20];
    const float* T_w1    = (const float*)d_in[21];
    const float* T_b1    = (const float*)d_in[22];
    const float* T_w2    = (const float*)d_in[23];
    const float* pred_g  = (const float*)d_in[24];
    const float* pred_b  = (const float*)d_in[25];
    const float* pred_w1 = (const float*)d_in[26];
    const float* pred_b1 = (const float*)d_in[27];
    const float* pred_w2 = (const float*)d_in[28];
    const float* pred_b2 = (const float*)d_in[29];
    const float* head_g  = (const float*)d_in[30];
    const float* head_b  = (const float*)d_in[31];
    const float* head_w  = (const float*)d_in[32];
    const float* head_bs = (const float*)d_in[33];
    float* out = (float*)d_out;

    // -------- workspace bump allocator (with reuse) --------
    char* base = (char*)d_ws; size_t off = 0;
    auto alloc = [&](size_t bytes) -> void* {
        void* p = base + off; off = (off + bytes + 255) & ~(size_t)255; return p;
    };
    // weights, all stored transposed (N,Kpad) bf16
    bf16* lin_wb   = (bf16*)alloc((size_t)DM * DINP * 2);
    bf16* enc_w1b  = (bf16*)alloc((size_t)DB * DM * 2);
    bf16* enc_w2b  = (bf16*)alloc((size_t)DM * DB * 2);
    bf16* K_wb     = (bf16*)alloc((size_t)DM * DM * 2);
    bf16* T_w1b    = (bf16*)alloc((size_t)DB * DM * 2);
    bf16* T_w2b    = (bf16*)alloc((size_t)DM * DB * 2);
    bf16* pred_w1b = (bf16*)alloc((size_t)DB * DM * 2);
    bf16* pred_w2b = (bf16*)alloc((size_t)DM * DB * 2);

    bf16*  featC  = (bf16*)alloc((size_t)NCAND * DINP * 2);  // reused: diff (BQ*CTX x DM)
    bf16*  featQ  = (bf16*)alloc((size_t)BQ * DINP * 2);
    float* xC     = (float*)alloc((size_t)NCAND * DM * 4);   // reused: values (BQ*CTX x DM)
    bf16*  xCb    = (bf16*)alloc((size_t)NCAND * DM * 2);    // reused: lnC
    bf16*  bufH   = (bf16*)alloc((size_t)NCAND * DB * 2);    // reused: dot f32, T-h bf16
    float* candk  = (float*)alloc((size_t)NCAND * DM * 4);
    bf16*  candkb = (bf16*)alloc((size_t)NCAND * DM * 2);
    float* cnorm  = (float*)alloc((size_t)NCAND * 4);
    float* qx     = (float*)alloc((size_t)BQ * DM * 4);
    bf16*  qxb    = (bf16*)alloc((size_t)BQ * DM * 2);       // reused: query LN, pred LN
    bf16*  qhb    = (bf16*)alloc((size_t)BQ * DB * 2);
    float* qk     = (float*)alloc((size_t)BQ * DM * 4);
    bf16*  qkb    = (bf16*)alloc((size_t)BQ * DM * 2);
    int*   ctx    = (int*)alloc((size_t)BQ * CTX * 4);
    float* sim    = (float*)alloc((size_t)BQ * CTX * 4);
    (void)ws_size; (void)n_in; (void)in_sizes; (void)out_size;

    float* dot    = (float*)bufH;   // 512*50000*4 == 50000*1024*2 bytes
    bf16*  diffb  = featC;
    float* values = xC;
    bf16*  thb    = bufH;

    auto gemm = [&](const bf16* A, const bf16* Bm, const float* bias, const float* res,
                    float* Cf, bf16* Cb, int M, int N, int K, int ldb, int relu) {
        dim3 g((N + 127) / 128, (M + 127) / 128);
        gemm_bf16_kernel<<<g, 256, 0, stream>>>(A, Bm, bias, res, Cf, Cb, M, N, K, ldb, relu);
    };
    auto cvt = [&](const float* s, bf16* d, int K, int N, int Kpad) {
        int n = N * Kpad;
        cvt_wT_kernel<<<(n + 255) / 256, 256, 0, stream>>>(s, d, K, N, Kpad);
    };

    // ---- weight conversion (f32 (K,N) -> bf16 (N,Kpad) transposed) ----
    cvt(lin_w,   lin_wb,   DIN, DM, DINP);
    cvt(enc_w1,  enc_w1b,  DM,  DB, DM);
    cvt(enc_w2,  enc_w2b,  DB,  DM, DB);
    cvt(K_w,     K_wb,     DM,  DM, DM);
    cvt(T_w1,    T_w1b,    DM,  DB, DM);
    cvt(T_w2,    T_w2b,    DB,  DM, DB);
    cvt(pred_w1, pred_w1b, DM,  DB, DM);
    cvt(pred_w2, pred_w2b, DB,  DM, DB);

    // ---- PLR features ----
    plr_kernel<<<NCAND, 256, 0, stream>>>(c_num, c_cat, freq, plr_w, plr_b, featC, NCAND);
    plr_kernel<<<BQ,    256, 0, stream>>>(x_num, x_cat, freq, plr_w, plr_b, featQ, BQ);

    // ---- candidate encode: lin -> enc MLP (+res) -> LN -> K ----
    gemm(featC, lin_wb, lin_b, nullptr, xC, xCb, NCAND, DM, DINP, DINP, 0);
    gemm(xCb, enc_w1b, enc_b1, nullptr, nullptr, bufH, NCAND, DB, DM, DM, 1);
    gemm(bufH, enc_w2b, enc_b2, xC, xC, nullptr, NCAND, DM, DB, DB, 0);
    ln_kernel<<<NCAND, 256, 0, stream>>>(xC, mix_g, mix_b, xCb);
    gemm(xCb, K_wb, K_b, nullptr, candk, candkb, NCAND, DM, DM, DM, 0);
    rownorm_kernel<<<NCAND, 256, 0, stream>>>(candk, cnorm);

    // ---- query encode ----
    gemm(featQ, lin_wb, lin_b, nullptr, qx, qxb, BQ, DM, DINP, DINP, 0);
    gemm(qxb, enc_w1b, enc_b1, nullptr, nullptr, qhb, BQ, DB, DM, DM, 1);
    gemm(qhb, enc_w2b, enc_b2, qx, qx, nullptr, BQ, DM, DB, DB, 0);
    ln_kernel<<<BQ, 256, 0, stream>>>(qx, mix_g, mix_b, qxb);
    gemm(qxb, K_wb, K_b, nullptr, qk, qkb, BQ, DM, DM, DM, 0);

    // ---- kNN: dot = qk @ candk^T, then per-row top-96 argmin of d2 ----
    gemm(qkb, candkb, nullptr, nullptr, dot, nullptr, BQ, NCAND, DM, DM, 0);
    topk_kernel<<<BQ, 256, 0, stream>>>(dot, cnorm, ctx, NCAND);

    // ---- context gather + T transform + softmax mix ----
    gather_kernel<<<BQ * CTX, 256, 0, stream>>>(qk, candk, ctx, c_y, label_w, label_b,
                                                diffb, values, sim);
    gemm(diffb, T_w1b, T_b1, nullptr, nullptr, thb, BQ * CTX, DB, DM, DM, 1);
    gemm(thb, T_w2b, nullptr, values, values, nullptr, BQ * CTX, DM, DB, DB, 0);
    mix_kernel<<<BQ, 256, 0, stream>>>(qx, sim, values);

    // ---- predictor block (prenorm) ----
    ln_kernel<<<BQ, 256, 0, stream>>>(qx, pred_g, pred_b, qxb);
    gemm(qxb, pred_w1b, pred_b1, nullptr, nullptr, qhb, BQ, DB, DM, DM, 1);
    gemm(qhb, pred_w2b, pred_b2, qx, qx, nullptr, BQ, DM, DB, DB, 0);

    // ---- head ----
    head_kernel<<<BQ, 256, 0, stream>>>(qx, head_g, head_b, head_w, head_bs, out);
}
